// SUR_73031623901809
// MI455X (gfx1250) — matrix operand order repaired and checked by reading, not compiled, not run on hardware
//
#include <hip/hip_runtime.h>

#define B_TOTAL 16384
#define DIM 128
#define ROWS 128              // batch rows per workgroup
#define KA 256                // A row length = 2*DIM
#define WPAD 8
#define WROW (KA + WPAD)      // 264 f16 elements per padded LDS row (528 B)

typedef __attribute__((ext_vector_type(16))) _Float16 v16h;
typedef __attribute__((ext_vector_type(8)))  float    v8f;

__device__ __forceinline__ float wred32(float v) {
  v += __shfl_xor(v, 1, 32);
  v += __shfl_xor(v, 2, 32);
  v += __shfl_xor(v, 4, 32);
  v += __shfl_xor(v, 8, 32);
  v += __shfl_xor(v, 16, 32);
  return v;
}
__device__ __forceinline__ float wred16(float v) {
  v += __shfl_xor(v, 1, 32);
  v += __shfl_xor(v, 2, 32);
  v += __shfl_xor(v, 4, 32);
  v += __shfl_xor(v, 8, 32);
  return v;
}
__device__ __forceinline__ float dot4(float4 a, float4 b) {
  return a.x * b.x + a.y * b.y + a.z * b.z + a.w * b.w;
}
__device__ __forceinline__ void st4h(_Float16* p, float4 v) {
  union { _Float16 h[4]; uint2 u; } t;
  t.h[0] = (_Float16)v.x; t.h[1] = (_Float16)v.y;
  t.h[2] = (_Float16)v.z; t.h[3] = (_Float16)v.w;
  *(uint2*)p = t.u;
}

extern "C" __global__ __launch_bounds__(256)
void sur_main(const int* __restrict__ h, const int* __restrict__ r,
              const int* __restrict__ pos_t, const int* __restrict__ neg_t,
              const float* __restrict__ entity, const float* __restrict__ relation,
              const float* __restrict__ h_w1, const float* __restrict__ h_w2,
              const float* __restrict__ h_b,
              const float* __restrict__ r_w1, const float* __restrict__ r_w2,
              const float* __restrict__ r_b,
              const float* __restrict__ W, float* __restrict__ acc)
{
  extern __shared__ _Float16 smem[];
  _Float16* sWT = smem;                // [DIM][WROW]  : W^T as f16 (n-major)
  _Float16* sA  = smem + DIM * WROW;   // [ROWS][WROW] : concat(cross_h, cross_r)

  const int tid    = threadIdx.x;
  const int wv     = tid >> 5;         // wave id 0..7
  const int ln     = tid & 31;         // lane  0..31
  const int wgBase = blockIdx.x * ROWS;

  // ---- Phase 0: stage sem_trans_w (256x128 f32, [k][n]) -> sWT[n][k] f16 ----
  for (int idx = tid; idx < KA * DIM; idx += 256) {
    int k = idx >> 7;
    int n = idx & (DIM - 1);
    sWT[n * WROW + k] = (_Float16)W[idx];
  }

  // ---- Phase 1: cross_h / cross_r for this WG's 128 rows -> sA (f16) ----
  const int c4 = ln * 4;
  const float4 vh1 = *(const float4*)(h_w1 + c4);   // h_trans_w1
  const float4 vh2 = *(const float4*)(h_w2 + c4);   // h_trans_w2
  const float4 vr1 = *(const float4*)(r_w1 + c4);   // r_trans_w1
  const float4 vr2 = *(const float4*)(r_w2 + c4);   // r_trans_w2
  const float4 bh  = *(const float4*)(h_b + c4);
  const float4 br  = *(const float4*)(r_b + c4);

  float l2c = 0.0f;
  for (int it = 0; it < 16; ++it) {
    int row = wv * 16 + it;
    int b   = wgBase + row;
    const float4 he = *(const float4*)(entity   + (size_t)h[b] * DIM + c4);
    const float4 re = *(const float4*)(relation + (size_t)r[b] * DIM + c4);
    // reference: r_w1 = r_e@h_trans_w1, r_w2 = r_e@h_trans_w2,
    //            h_w1 = h_e@r_trans_w1, h_w2 = h_e@r_trans_w2
    float hw1 = wred32(dot4(he, vr1));
    float hw2 = wred32(dot4(he, vr2));
    float rw1 = wred32(dot4(re, vh1));
    float rw2 = wred32(dot4(re, vh2));
    float4 ch, cr;
    ch.x = he.x * rw1 + re.x * hw2 + bh.x;
    ch.y = he.y * rw1 + re.y * hw2 + bh.y;
    ch.z = he.z * rw1 + re.z * hw2 + bh.z;
    ch.w = he.w * rw1 + re.w * hw2 + bh.w;
    cr.x = he.x * rw2 + re.x * hw1 + br.x;
    cr.y = he.y * rw2 + re.y * hw1 + br.y;
    cr.z = he.z * rw2 + re.z * hw1 + br.z;
    cr.w = he.w * rw2 + re.w * hw1 + br.w;
    l2c += 0.5f * (ch.x*ch.x + ch.y*ch.y + ch.z*ch.z + ch.w*ch.w +
                   cr.x*cr.x + cr.y*cr.y + cr.z*cr.z + cr.w*cr.w);
    _Float16* ap = sA + row * WROW;
    st4h(ap + c4, ch);
    st4h(ap + DIM + c4, cr);
  }
  __syncthreads();

  // ---- Phase 2: pred_t tile (16 rows x 128 cols) via v_wmma_f32_16x16x32_f16 ----
  const int hf = ln >> 4;      // half-wave
  const int lm = ln & 15;
  v8f accv[8] = {};
  const _Float16* aBase = sA + (wv * 16 + lm) * WROW;
  #pragma unroll
  for (int kt = 0; kt < 8; ++kt) {
    // A frag (16x32 f16): half0 holds K 0-7 & 16-23, half1 holds K 8-15 & 24-31
    union { v16h v; uint4 u[2]; } fa;
    const _Float16* ap = aBase + kt * 32 + hf * 8;
    fa.u[0] = *(const uint4*)(ap);
    fa.u[1] = *(const uint4*)(ap + 16);
    #pragma unroll
    for (int t = 0; t < 8; ++t) {
      // B frag (32x16 f16): lane = col N; half0 K 0-7 & 16-23, half1 K 8-15 & 24-31
      union { v16h v; uint4 u[2]; } fb;
      const _Float16* bp = sWT + (t * 16 + lm) * WROW + kt * 32 + hf * 8;
      fb.u[0] = *(const uint4*)(bp);
      fb.u[1] = *(const uint4*)(bp + 16);
      accv[t] = __builtin_amdgcn_wmma_f32_16x16x32_f16(
          false, fa.v, false, fb.v, (short)0, accv[t], false, false);
    }
  }

  // ---- Phase 3: scores + softplus + L2 from the C-fragment layout ----
  // C layout: lane = N (lm), VGPR r = row M = r + 8*half
  float l2e   = 0.0f;
  float lossw = 0.0f;
  #pragma unroll
  for (int rr = 0; rr < 8; ++rr) {
    int b = wgBase + wv * 16 + hf * 8 + rr;
    const float* pe = entity + (size_t)pos_t[b] * DIM + lm;
    const float* ne = entity + (size_t)neg_t[b] * DIM + lm;
    float ps = 0.0f, ns = 0.0f;
    #pragma unroll
    for (int t = 0; t < 8; ++t) {
      float c  = accv[t][rr];
      float pv = pe[t * 16];
      float nv = ne[t * 16];
      ps += c * pv;
      ns += c * nv;
      l2e += 0.5f * (pv * pv + nv * nv);
    }
    ps = wred16(ps);                       // uniform across half-wave
    ns = wred16(ns);
    float x  = ns - ps;                    // -(pos - neg)
    float sp = fmaxf(x, 0.0f) + log1pf(expf(-fabsf(x)));   // stable softplus
    lossw += sp;
  }
  lossw += __shfl_xor(lossw, 16, 32);      // combine halves (uniform within half)
  float l2tot = wred32(l2c + l2e);
  if (ln == 0) {
    atomicAdd(&acc[0], lossw);
    atomicAdd(&acc[1], l2tot);
  }
}

extern "C" __global__ void sur_fin(const float* __restrict__ acc,
                                   float* __restrict__ out) {
  out[0] = acc[0] * (1.0f / B_TOTAL) + 1e-5f * (acc[1] * (1.0f / B_TOTAL));
}

extern "C" void kernel_launch(void* const* d_in, const int* in_sizes, int n_in,
                              void* d_out, int out_size, void* d_ws, size_t ws_size,
                              hipStream_t stream)
{
  const int*   h        = (const int*)d_in[0];
  const int*   r        = (const int*)d_in[1];
  const int*   pos_t    = (const int*)d_in[2];
  const int*   neg_t    = (const int*)d_in[3];
  const float* entity   = (const float*)d_in[4];
  const float* relation = (const float*)d_in[5];
  const float* h_w1     = (const float*)d_in[6];
  const float* h_w2     = (const float*)d_in[7];
  const float* h_b      = (const float*)d_in[8];
  const float* r_w1     = (const float*)d_in[9];
  const float* r_w2     = (const float*)d_in[10];
  const float* r_b      = (const float*)d_in[11];
  const float* W        = (const float*)d_in[12];

  float* acc = (float*)d_ws;
  hipMemsetAsync(acc, 0, 2 * sizeof(float), stream);

  size_t shmem = (size_t)(DIM + ROWS) * WROW * sizeof(_Float16);  // 132 KB < 320 KB/WGP
  sur_main<<<dim3(B_TOTAL / ROWS), dim3(256), shmem, stream>>>(
      h, r, pos_t, neg_t, entity, relation,
      h_w1, h_w2, h_b, r_w1, r_w2, r_b, W, acc);
  sur_fin<<<1, 1, 0, stream>>>(acc, (float*)d_out);
}
